// Net_54872502174330
// MI455X (gfx1250) — matrix operand order repaired and checked by reading, not compiled, not run on hardware
//
#include <hip/hip_runtime.h>
#include <math.h>

// ---------------------------------------------------------------------------
// GCN (2-layer) for MI455X / gfx1250.
//   Xbf   = bf16(x); W1T/W2T = bf16 transposed weights      (pre-pass)
//   h1    = Xbf @ W1          (v_wmma_f32_16x16x32_bf16, b128 fragment loads)
//   agg1  = scatter_add(norm * h1[src]) ; hrelu = bf16(relu(agg1 + b1))
//   h2    = hrelu @ W2        (WMMA)
//   out   = scatter_add(norm * h2[src]) + b2 ; log_softmax
// ---------------------------------------------------------------------------

typedef __attribute__((ext_vector_type(16))) __bf16        v16bf;
typedef __attribute__((ext_vector_type(8)))  float         v8f;
typedef __attribute__((ext_vector_type(4)))  unsigned int  v4u;

union Frag {
  v16bf v;
  v4u   q[2];
};

__device__ __forceinline__ unsigned short f2bf(float f) {
  unsigned int u = __float_as_uint(f);
  u += 0x7fffu + ((u >> 16) & 1u);      // round-to-nearest-even
  return (unsigned short)(u >> 16);
}

__device__ __forceinline__ unsigned int pk2bf(float lo, float hi) {
  return (unsigned int)f2bf(lo) | ((unsigned int)f2bf(hi) << 16);
}

// ---------------- bf16 conversion pre-passes --------------------------------

// x [N,256] f32 -> bf16, 8 elements per thread (2x float4 in, 1x uint4 out)
__global__ void cvt_x_kernel(const float* __restrict__ X, unsigned int* __restrict__ Xbf,
                             long long total8) {
  long long t = (long long)blockIdx.x * blockDim.x + threadIdx.x;
  if (t >= total8) return;
  const float4 a = *(const float4*)(X + t * 8);
  const float4 b = *(const float4*)(X + t * 8 + 4);
  v4u o;
  o.x = pk2bf(a.x, a.y);
  o.y = pk2bf(a.z, a.w);
  o.z = pk2bf(b.x, b.y);
  o.w = pk2bf(b.z, b.w);
  *(v4u*)(Xbf + t * 4) = o;
}

// W [K,Ncol] f32 row-major -> WT [Ncol,K] bf16 (each output col contiguous in K)
template <int K, int NCOL>
__global__ void cvt_wt_kernel(const float* __restrict__ W, unsigned short* __restrict__ WT) {
  int t = blockIdx.x * blockDim.x + threadIdx.x;
  if (t >= K * NCOL) return;
  const int n = t / K;
  const int k = t % K;
  WT[t] = f2bf(W[(size_t)k * NCOL + n]);
}

// ---------------- degree + normalization -----------------------------------

__global__ void deg_kernel(const int* __restrict__ ei, float* __restrict__ deg,
                           int E, int N) {
  int t = blockIdx.x * blockDim.x + threadIdx.x;
  if (t >= E + N) return;
  int d = (t < E) ? ei[E + t] : (t - E);   // dst half of edge_index, then self-loops
  unsafeAtomicAdd(&deg[d], 1.0f);
}

__global__ void dinv_kernel(float* __restrict__ deg, int N) {
  int i = blockIdx.x * blockDim.x + threadIdx.x;
  if (i >= N) return;
  float v = deg[i];
  deg[i] = (v > 0.0f) ? rsqrtf(v) : 0.0f;
}

// ---------------- WMMA GEMMs -------------------------------------------------
// Fragment layouts per CDNA5 ISA 7.12.2:
//   A 16-bit 16x32: lane half h, VGPR v -> K = (v&3)*2 + (v>>2)*16 + h*8 (+1)
//     => per lane two contiguous 8-element (16B) runs: [k0+8h .. +7], [k0+16+8h .. +7]
//   B 16-bit 32x16: lane n=l&15, half h -> K = [k0+8h .. +7], [k0+16+8h .. +7] (same as A)
//   C/D f32: m = v + 8h, n = l&15

// GEMM1: [N,256]bf16 x W1T[64,256]bf16 -> h1 [N,64] f32. 4 waves/block (H=64).
__global__ void gemm1_kernel(const unsigned short* __restrict__ Xbf,
                             const unsigned short* __restrict__ W1T,
                             float* __restrict__ Hout, int N) {
  const int lane  = threadIdx.x & 31;
  const int wave  = threadIdx.x >> 5;          // 0..3
  const int mBase = blockIdx.x * 16;
  const int nBase = wave * 16;
  const int half  = lane >> 4;
  const int ncol  = lane & 15;
  int arow = mBase + (lane & 15);
  if (arow >= N) arow = N - 1;                 // clamp (store is guarded)

  const unsigned short* ap0 = Xbf + (size_t)arow * 256 + half * 8;
  const unsigned short* bp0 = W1T + (size_t)(nBase + ncol) * 256 + half * 8;

  v8f acc = {};
#pragma unroll
  for (int k0 = 0; k0 < 256; k0 += 32) {
    Frag a, b;
    a.q[0] = *(const v4u*)(ap0 + k0);
    a.q[1] = *(const v4u*)(ap0 + k0 + 16);
    b.q[0] = *(const v4u*)(bp0 + k0);
    b.q[1] = *(const v4u*)(bp0 + k0 + 16);
    acc = __builtin_amdgcn_wmma_f32_16x16x32_bf16(false, a.v, false, b.v,
                                                  (short)0, acc, false, false);
  }
#pragma unroll
  for (int v = 0; v < 8; ++v) {
    const int m = mBase + v + 8 * half;
    if (m < N) Hout[(size_t)m * 64 + nBase + ncol] = acc[v];
  }
}

// GEMM2: [N,64]bf16 x W2T[16,64]bf16 -> h2 [N,16] f32. 8 waves/block.
__global__ void gemm2_kernel(const unsigned short* __restrict__ Hb,
                             const unsigned short* __restrict__ W2T,
                             float* __restrict__ Out, int N) {
  const int lane  = threadIdx.x & 31;
  const int wave  = threadIdx.x >> 5;          // 0..7
  const int tiles = (N + 15) >> 4;
  const int tile  = blockIdx.x * 8 + wave;
  if (tile >= tiles) return;                   // uniform per wave: EXEC stays all-1s
  const int mBase = tile * 16;
  const int half  = lane >> 4;
  const int ncol  = lane & 15;
  int arow = mBase + (lane & 15);
  if (arow >= N) arow = N - 1;

  const unsigned short* ap0 = Hb + (size_t)arow * 64 + half * 8;
  const unsigned short* bp0 = W2T + (size_t)ncol * 64 + half * 8;

  v8f acc = {};
#pragma unroll
  for (int k0 = 0; k0 < 64; k0 += 32) {
    Frag a, b;
    a.q[0] = *(const v4u*)(ap0 + k0);
    a.q[1] = *(const v4u*)(ap0 + k0 + 16);
    b.q[0] = *(const v4u*)(bp0 + k0);
    b.q[1] = *(const v4u*)(bp0 + k0 + 16);
    acc = __builtin_amdgcn_wmma_f32_16x16x32_bf16(false, a.v, false, b.v,
                                                  (short)0, acc, false, false);
  }
#pragma unroll
  for (int v = 0; v < 8; ++v) {
    const int m = mBase + v + 8 * half;
    if (m < N) Out[(size_t)m * 16 + ncol] = acc[v];
  }
}

// ---------------- edge scatter: Agg[dst] += norm * Hsrc[src] -----------------
// F floats per node, F/4 threads per edge, float4 gathers + 4 f32 atomics.

template <int F, int LOG_TPE>
__global__ void scatter_kernel(const int* __restrict__ ei, const float* __restrict__ dinv,
                               const float* __restrict__ Hsrc, float* __restrict__ Agg,
                               int E, int N) {
  const long long tid = (long long)blockIdx.x * blockDim.x + threadIdx.x;
  const long long e   = tid >> LOG_TPE;
  const int       q   = (int)(tid & ((1 << LOG_TPE) - 1));
  if (e >= (long long)E + N) return;
  int s, d;
  if (e < E) { s = ei[e]; d = ei[E + e]; }
  else       { s = d = (int)(e - E); }
  const float w = dinv[s] * dinv[d];
  const float4 hv = *(const float4*)(Hsrc + (size_t)s * F + q * 4);
  float* o = Agg + (size_t)d * F + q * 4;
  unsafeAtomicAdd(o + 0, hv.x * w);
  unsafeAtomicAdd(o + 1, hv.y * w);
  unsafeAtomicAdd(o + 2, hv.z * w);
  unsafeAtomicAdd(o + 3, hv.w * w);
}

// ---------------- elementwise epilogues -------------------------------------

// hrelu_bf16[i] = bf16(max(agg1[i] + b1[i%64], 0)), 2 elements per thread
__global__ void relu_bias_bf_kernel(const float* __restrict__ A, const float* __restrict__ b,
                                    unsigned int* __restrict__ Out, int totalPairs) {
  int i = blockIdx.x * blockDim.x + threadIdx.x;
  if (i >= totalPairs) return;
  const int i2 = i * 2;
  float v0 = A[i2]     + b[i2 & 63];
  float v1 = A[i2 + 1] + b[(i2 + 1) & 63];
  v0 = (v0 > 0.0f) ? v0 : 0.0f;
  v1 = (v1 > 0.0f) ? v1 : 0.0f;
  Out[i] = pk2bf(v0, v1);
}

__global__ void logsoftmax_kernel(float* __restrict__ Out, const float* __restrict__ b2,
                                  int N) {
  int i = blockIdx.x * blockDim.x + threadIdx.x;
  if (i >= N) return;
  float* row = Out + (size_t)i * 16;
  float v[16];
  float m = -INFINITY;
#pragma unroll
  for (int c = 0; c < 16; ++c) {
    v[c] = row[c] + b2[c];
    m = fmaxf(m, v[c]);
  }
  float sum = 0.0f;
#pragma unroll
  for (int c = 0; c < 16; ++c) sum += expf(v[c] - m);
  const float lse = logf(sum) + m;
#pragma unroll
  for (int c = 0; c < 16; ++c) row[c] = v[c] - lse;
}

// ---------------------------------------------------------------------------

extern "C" void kernel_launch(void* const* d_in, const int* in_sizes, int n_in,
                              void* d_out, int out_size, void* d_ws, size_t ws_size,
                              hipStream_t stream) {
  const float* x  = (const float*)d_in[0];
  const int*   ei = (const int*)d_in[1];
  const float* W1 = (const float*)d_in[2];
  const float* b1 = (const float*)d_in[3];
  const float* W2 = (const float*)d_in[4];
  const float* b2 = (const float*)d_in[5];

  const int N  = in_sizes[0] / 256;   // F_IN = 256
  const int E  = in_sizes[1] / 2;
  const int EN = E + N;

  // --- workspace carve (256B aligned) ---------------------------------------
  char*  ws  = (char*)d_ws;
  size_t off = 0;
  auto carve = [&](size_t bytes) {
    void* p = ws + off;
    off += (bytes + 255) & ~(size_t)255;
    return p;
  };
  float*          deg = (float*)carve((size_t)N * 4);          // becomes dinv
  float*          h1  = (float*)carve((size_t)N * 64 * 4);     // 25.6 MB
  unsigned short* W1T = (unsigned short*)carve(256 * 64 * 2);
  unsigned short* W2T = (unsigned short*)carve(64 * 16 * 2);
  // Aliased region R (51.2 MB): Xbf during phase A; agg1/hrelu/h2 during phase B.
  char* R = (char*)carve((size_t)N * 256 * 2);
  unsigned short* Xbf   = (unsigned short*)R;                  // [N,256] bf16
  float*          agg1  = (float*)R;                           // [N,64]  f32
  size_t aoff = ((size_t)N * 64 * 4 + 255) & ~(size_t)255;
  unsigned short* hrelu = (unsigned short*)(R + aoff);         // [N,64]  bf16
  size_t hoff = aoff + (((size_t)N * 64 * 2 + 255) & ~(size_t)255);
  float*          h2    = (float*)(R + hoff);                  // [N,16]  f32
  float*          outp  = (float*)d_out;

  // --- phase A: conversions + degree ---------------------------------------
  hipMemsetAsync(deg,  0, (size_t)N * 4, stream);
  hipMemsetAsync(outp, 0, (size_t)out_size * 4, stream);

  {
    long long total8 = (long long)N * 256 / 8;
    cvt_x_kernel<<<(int)((total8 + 255) / 256), 256, 0, stream>>>(x, (unsigned int*)Xbf, total8);
  }
  cvt_wt_kernel<256, 64><<<(256 * 64 + 255) / 256, 256, 0, stream>>>(W1, W1T);
  cvt_wt_kernel<64, 16><<<(64 * 16 + 255) / 256, 256, 0, stream>>>(W2, W2T);

  deg_kernel<<<(EN + 255) / 256, 256, 0, stream>>>(ei, deg, E, N);
  dinv_kernel<<<(N + 255) / 256, 256, 0, stream>>>(deg, N);

  // h1 = x @ W1 (WMMA; last consumer of Xbf)
  gemm1_kernel<<<(N + 15) / 16, 128, 0, stream>>>(Xbf, W1T, h1, N);

  // --- phase B: R is reused for agg1/hrelu/h2 -------------------------------
  hipMemsetAsync(agg1, 0, (size_t)N * 64 * 4, stream);   // after gemm1 in stream order

  {
    long long threads = (long long)EN * 16;
    scatter_kernel<64, 4><<<(int)((threads + 255) / 256), 256, 0, stream>>>(
        ei, deg, h1, agg1, E, N);
  }

  relu_bias_bf_kernel<<<(N * 32 + 255) / 256, 256, 0, stream>>>(
      agg1, b1, (unsigned int*)hrelu, N * 32);

  {
    int tiles  = (N + 15) / 16;
    int blocks = (tiles + 7) / 8;
    gemm2_kernel<<<blocks, 256, 0, stream>>>(hrelu, W2T, h2, N);
  }

  {
    long long threads = (long long)EN * 4;
    scatter_kernel<16, 2><<<(int)((threads + 255) / 256), 256, 0, stream>>>(
        ei, deg, h2, outp, E, N);
  }

  logsoftmax_kernel<<<(N + 255) / 256, 256, 0, stream>>>(outp, b2, N);
}
